// VanillaRNNModel_63771674411087
// MI455X (gfx1250) — compile-verified, matrix operand order checked by reading, not run on hardware
//
#include <hip/hip_runtime.h>
#include <hip/hip_bf16.h>

typedef __attribute__((ext_vector_type(16))) _Float16 v16h;
typedef __attribute__((ext_vector_type(8)))  _Float16 v8h;
typedef __attribute__((ext_vector_type(8)))  float    v8f;

#define VS  128
#define EMB 64
#define HS  512
#define BSZ 256
#define TT  512

// workspace layout (units: f16 halves)
#define OFF_EMB   0
#define OFF_WIH0  (OFF_EMB  + VS*EMB)
#define OFF_WHH0  (OFF_WIH0 + EMB*HS)
#define OFF_WIH1  (OFF_WHH0 + HS*HS)
#define OFF_WHH1  (OFF_WIH1 + HS*HS)
#define OFF_WFC   (OFF_WHH1 + HS*HS)
// total = 892928 halves = ~1.79 MB (resident in 192MB L2)

#if __has_builtin(__builtin_amdgcn_tanhf)
  #define TANH(v) __builtin_amdgcn_tanhf(v)
#elif __has_builtin(__builtin_amdgcn_tanh_f32)
  #define TANH(v) __builtin_amdgcn_tanh_f32(v)
#else
  #define TANH(v) tanhf(v)
#endif

#define AROW (EMB + HS + HS)   // 1088 halves per activation row

// ---------------------------------------------------------------------------
__global__ void cvt_f16_kernel(const float* __restrict__ src,
                               _Float16* __restrict__ dst, int n) {
  int i = blockIdx.x * blockDim.x + threadIdx.x;
  if (i < n) dst[i] = (_Float16)src[i];
}

// Pack [K,N] row-major fp32 weight into WMMA-B fragment order, f16.
__global__ void pack_b_kernel(const float* __restrict__ src,
                              _Float16* __restrict__ dst, int K, int N,
                              int ktShift) {
  int idx = blockIdx.x * blockDim.x + threadIdx.x;
  if (idx >= K * N) return;
  int f    = idx >> 9;
  int w    = idx & 511;
  int l    = w >> 4;
  int h    = w & 15;
  int j    = f >> ktShift;
  int kk   = f & ((1 << ktShift) - 1);
  int col  = j * 16 + (l & 15);
  int krow = kk * 32 + (h & 7) + ((h >> 3) << 4) + ((l >> 4) << 3);
  dst[idx] = (_Float16)src[krow * N + col];
}

// ---------------------------------------------------------------------------
__device__ __forceinline__ v16h load_a(const _Float16* ap) {
  v8h lo = *(const v8h*)ap;
  v8h hi = *(const v8h*)(ap + 16);
  return __builtin_shufflevector(lo, hi, 0,1,2,3,4,5,6,7,8,9,10,11,12,13,14,15);
}

__device__ __forceinline__ v8f wmma_step(const _Float16* ap,
                                         const _Float16* bp, v8f c) {
  v16h a = load_a(ap);
  v16h b = *(const v16h*)bp;
  return __builtin_amdgcn_wmma_f32_16x16x32_f16(false, a, false, b,
                                                (short)0, c, false, false);
}

__device__ __forceinline__ v8f wmma_step_breg(const _Float16* ap,
                                              v16h b, v8f c) {
  v16h a = load_a(ap);
  return __builtin_amdgcn_wmma_f32_16x16x32_f16(false, a, false, b,
                                                (short)0, c, false, false);
}

// Fused persistent RNN + FC kernel.
// grid = BSZ/16 blocks, 1024 threads (32 wave32 waves).
__global__ __launch_bounds__(1024, 1)
void rnn_fused_kernel(const int* __restrict__ x,
                      const _Float16* __restrict__ ws,
                      const float* __restrict__ bih0, const float* __restrict__ bhh0,
                      const float* __restrict__ bih1, const float* __restrict__ bhh1,
                      const float* __restrict__ bfc,
                      float* __restrict__ out) {
  __shared__ _Float16     act[16][AROW]; // [emb|h0|h1] x 16 rows = 34 KB
  __shared__ float        lfc[16][VS];   // FC reduction buffer   =  8 KB
  __shared__ unsigned char tok[16][TT];  // token slab            =  8 KB

  const int tid  = threadIdx.x;
  const int lane = tid & 31;
  const int wv   = tid >> 5;
  const int bm   = blockIdx.x;

  // zero initial hidden state + stage this block's tokens (coalesced)
  for (int s = tid; s < 16 * (HS + HS); s += 1024)
    act[s >> 10][EMB + (s & 1023)] = (_Float16)0.0f;
  for (int s = tid; s < 16 * TT; s += 1024)
    tok[s >> 9][s & 511] = (unsigned char)x[(bm * 16 + (s >> 9)) * TT + (s & 511)];

  const int m    = lane & 15;
  const int khi  = (lane >> 4) << 3;
  const int mhi  = (lane >> 4) << 3;
  const int ncol = lane & 15;

  // per-wave weight fragment base pointers
  const size_t lb = (size_t)lane * 16;
  const _Float16* Bih0 = ws + OFF_WIH0 + (((size_t)wv * 2)  << 9) + lb;
  const _Float16* Bhh0 = ws + OFF_WHH0 + (((size_t)wv * 16) << 9) + lb;
  const _Float16* Bih1 = ws + OFF_WIH1 + (((size_t)wv * 16) << 9) + lb;
  const _Float16* Bhh1 = ws + OFF_WHH1 + (((size_t)wv * 16) << 9) + lb;
  const int fcN = wv & 7, fcK = wv >> 3;
  const _Float16* Bfc  = ws + OFF_WFC +
      (((size_t)(fcN * 16 + fcK * 4)) << 9) + lb;
  const _Float16* emb16 = ws + OFF_EMB;

  // t-invariant Wih0 fragments: keep resident in VGPRs for the whole kernel
  const v16h bih0f0 = *(const v16h*)(Bih0);
  const v16h bih0f1 = *(const v16h*)(Bih0 + 512);

  // biases (loaded once)
  const int nc = wv * 16 + ncol;
  const float bias0  = bih0[nc] + bhh0[nc];
  const float bias1  = bih1[nc] + bhh1[nc];
  const float biasfc = bfc[tid & 127];
  const int   lrow   = tid >> 7;
  const int   lcol   = tid & 127;

  const _Float16* arow = &act[m][khi];
  _Float16* h0w = &act[mhi][EMB + wv * 16 + ncol];
  _Float16* h1w = h0w + HS;

  const int er = tid >> 6, ec = tid & 63;   // embedding stage row/col

  __syncthreads();   // tokens staged (needed before first prefetch)

  // software-pipelined embedding element: value for step t is loaded during
  // step t-1's FC phase (hidden under WMMAs), so the step head is just an
  // LDS store.
  _Float16 ev = emb16[(int)tok[er][0] * EMB + ec];

  for (int t = 0; t < TT; ++t) {
    // ---- stage prefetched embedding + bias-init FC buffer ---------------
    act[er][ec] = ev;
    lfc[lrow][lcol]     = biasfc;
    lfc[lrow + 8][lcol] = biasfc;
    __syncthreads();

    // ---- layer 0: h0 = tanh([emb|h0] @ [Wih0;Whh0] + b), K = 576 --------
    v8f c0;
    #pragma unroll
    for (int i = 0; i < 8; ++i) c0[i] = bias0;
    c0 = wmma_step_breg(arow,      bih0f0, c0);
    c0 = wmma_step_breg(arow + 32, bih0f1, c0);
    #pragma unroll 2
    for (int kt = 0; kt < 16; ++kt)
      c0 = wmma_step(arow + EMB + kt * 32, Bhh0 + kt * 512, c0);
    __syncthreads();                           // all layer-0 A reads done
    #pragma unroll
    for (int i = 0; i < 8; ++i) h0w[i * AROW] = (_Float16)TANH(c0[i]);
    __syncthreads();

    // ---- layer 1: h1 = tanh([h0|h1] @ [Wih1;Whh1] + b), K = 1024 --------
    v8f c1;
    #pragma unroll
    for (int i = 0; i < 8; ++i) c1[i] = bias1;
    #pragma unroll 2
    for (int kt = 0; kt < 16; ++kt)
      c1 = wmma_step(arow + EMB + kt * 32, Bih1 + kt * 512, c1);
    #pragma unroll 2
    for (int kt = 0; kt < 16; ++kt)
      c1 = wmma_step(arow + EMB + HS + kt * 32, Bhh1 + kt * 512, c1);
    __syncthreads();                           // all layer-1 A reads done
    #pragma unroll
    for (int i = 0; i < 8; ++i) h1w[i * AROW] = (_Float16)TANH(c1[i]);
    __syncthreads();                           // h1 visible to FC readers

    // ---- prefetch next step's embedding element (hidden under FC) -------
    if (t + 1 < TT)
      ev = emb16[(int)tok[er][t + 1] * EMB + ec];

    // ---- FC: logits = h1 @ Wfc + bfc, K split 4-ways over all waves -----
    v8f cf;
    #pragma unroll
    for (int i = 0; i < 8; ++i) cf[i] = 0.0f;
    #pragma unroll 2
    for (int kq = 0; kq < 4; ++kq)
      cf = wmma_step(arow + EMB + HS + (fcK * 4 + kq) * 32,
                     Bfc + kq * 512, cf);
    #pragma unroll
    for (int i = 0; i < 8; ++i)
      atomicAdd(&lfc[i + mhi][fcN * 16 + ncol], cf[i]);   // ds_add_f32
    __syncthreads();                           // partials reduced

    // coalesced logits store (each thread owns its lfc elements)
    out[((size_t)(bm * 16 + lrow)     * TT + t) * VS + lcol] = lfc[lrow][lcol];
    out[((size_t)(bm * 16 + lrow + 8) * TT + t) * VS + lcol] = lfc[lrow + 8][lcol];
  }
}

// ---------------------------------------------------------------------------
extern "C" void kernel_launch(void* const* d_in, const int* in_sizes, int n_in,
                              void* d_out, int out_size, void* d_ws, size_t ws_size,
                              hipStream_t stream) {
  (void)in_sizes; (void)n_in; (void)out_size; (void)ws_size;
  const int*   x    = (const int*)  d_in[0];
  const float* emb  = (const float*)d_in[1];
  const float* Wih0 = (const float*)d_in[2];
  const float* bih0 = (const float*)d_in[3];
  const float* Whh0 = (const float*)d_in[4];
  const float* bhh0 = (const float*)d_in[5];
  const float* Wih1 = (const float*)d_in[6];
  const float* bih1 = (const float*)d_in[7];
  const float* Whh1 = (const float*)d_in[8];
  const float* bhh1 = (const float*)d_in[9];
  const float* Wfc  = (const float*)d_in[10];
  const float* bfc  = (const float*)d_in[11];
  _Float16* ws = (_Float16*)d_ws;

  cvt_f16_kernel<<<(VS*EMB + 255) / 256, 256, 0, stream>>>(emb, ws + OFF_EMB, VS*EMB);
  pack_b_kernel<<<(EMB*HS + 255) / 256, 256, 0, stream>>>(Wih0, ws + OFF_WIH0, EMB, HS, 1);
  pack_b_kernel<<<(HS*HS  + 255) / 256, 256, 0, stream>>>(Whh0, ws + OFF_WHH0, HS, HS, 4);
  pack_b_kernel<<<(HS*HS  + 255) / 256, 256, 0, stream>>>(Wih1, ws + OFF_WIH1, HS, HS, 4);
  pack_b_kernel<<<(HS*HS  + 255) / 256, 256, 0, stream>>>(Whh1, ws + OFF_WHH1, HS, HS, 4);
  pack_b_kernel<<<(HS*VS  + 255) / 256, 256, 0, stream>>>(Wfc,  ws + OFF_WFC,  HS, VS, 4);

  rnn_fused_kernel<<<BSZ / 16, 1024, 0, stream>>>(
      x, ws, bih0, bhh0, bih1, bhh1, bfc, (float*)d_out);
}
